// MultiHeadAttention_77678778515656
// MI455X (gfx1250) — matrix operand order repaired and checked by reading, not compiled, not run on hardware
//
#include <hip/hip_runtime.h>
#include <hip/hip_bf16.h>

typedef __attribute__((ext_vector_type(16))) _Float16 v16h;
typedef __attribute__((ext_vector_type(8)))  _Float16 v8h;
typedef __attribute__((ext_vector_type(8)))  float    v8f;

constexpr int B_   = 4;
constexpr int S_   = 2048;
constexpr int HID_ = 512;
constexpr int NH_  = 8;
constexpr int DH_  = 64;
constexpr int MROWS = B_ * S_;               // 8192
constexpr float NEGC = -1000000.0f;

__device__ __forceinline__ v8f wmma_f16(v16h a, v16h b, v8f c) {
  // D = A(16x32 f16) * B(32x16 f16) + C(16x16 f32)
  return __builtin_amdgcn_wmma_f32_16x16x32_f16(
      /*neg_a=*/false, a, /*neg_b=*/false, b,
      /*c_mod=*/(short)0, c, /*reuse_a=*/false, /*reuse_b=*/false);
}

__device__ __forceinline__ v16h cat8(v8h lo, v8h hi) {
  return __builtin_shufflevector(lo, hi, 0, 1, 2, 3, 4, 5, 6, 7,
                                 8, 9, 10, 11, 12, 13, 14, 15);
}

// ---------------------------------------------------------------------------
// f32 -> f16 elementwise convert
// ---------------------------------------------------------------------------
__global__ void k_cvt(const float* __restrict__ src, _Float16* __restrict__ dst, int n) {
  int i = blockIdx.x * blockDim.x + threadIdx.x;
  if (i < n) dst[i] = (_Float16)src[i];
}

// ---------------------------------------------------------------------------
// Generic NT GEMM:  C[n][o] = sum_i A[n][i] * W[o][i]
//   A: MROWS x 512 f16 row-major, W: 512 x 512 f16 row-major
//   mode 0: f16 out, * scale, layout (b,h,s,d)   (Q projection, scale = 1/8)
//   mode 1: f16 out, layout (b,h,s,d)            (K projection)
//   mode 2: f16 out, layout (b,h,d,s)            (V projection, transposed)
//   mode 3: f32 out, row-major (n,o)             (final output projection)
// Block: 128 threads (4 waves), each wave owns a 16n x 64o tile.
// ---------------------------------------------------------------------------
__global__ void k_gemm_xwt(const _Float16* __restrict__ A,
                           const _Float16* __restrict__ W,
                           int mode, float scale,
                           _Float16* __restrict__ outh,
                           float* __restrict__ outf) {
  const int lane = threadIdx.x & 31;
  const int wid  = threadIdx.x >> 5;
  const int l16  = lane & 15;
  const int hlf  = lane >> 4;
  const int n0 = blockIdx.x * 64 + wid * 16;
  const int o0 = blockIdx.y * 64;

  v8f acc[4];
#pragma unroll
  for (int t = 0; t < 4; ++t) acc[t] = (v8f){};

  const _Float16* arow = A + (size_t)(n0 + l16) * HID_;
#pragma unroll 4
  for (int k0 = 0; k0 < HID_; k0 += 32) {
    // A-layout 16x32 f16: contiguous chunk [16*hlf, +16)
    v8h lo = *(const v8h*)(arow + k0 + 16 * hlf);
    v8h hi = *(const v8h*)(arow + k0 + 16 * hlf + 8);
    v16h a = cat8(lo, hi);
#pragma unroll
    for (int t = 0; t < 4; ++t) {
      // B-layout 32x16: lane holds N = o col, K contiguous (32B load)
      v16h b = *(const v16h*)(W + (size_t)(o0 + t * 16 + l16) * HID_ + k0 + 16 * hlf);
      acc[t] = wmma_f16(a, b, acc[t]);
    }
  }

#pragma unroll
  for (int t = 0; t < 4; ++t) {
#pragma unroll
    for (int r = 0; r < 8; ++r) {
      const int n = n0 + r + 8 * hlf;          // C/D layout: row = r + 8*half
      const int o = o0 + t * 16 + l16;         // col = lane & 15
      const float v = acc[t][r] * scale;
      if (mode == 3) {
        outf[(size_t)n * HID_ + o] = v;
      } else {
        const int b = n >> 11, s = n & (S_ - 1);
        const int h = o >> 6,  d = o & (DH_ - 1);
        if (mode == 2)
          outh[(((size_t)(b * NH_ + h)) * DH_ + d) * S_ + s] = (_Float16)v;
        else
          outh[(((size_t)(b * NH_ + h)) * S_ + s) * DH_ + d] = (_Float16)v;
      }
    }
  }
}

// ---------------------------------------------------------------------------
// Attention core: one wave (32 threads) per 16-query tile.
//   Qp, Kp: (b,h,s,d) f16 (Q pre-scaled by 1/sqrt(DH));  Vt: (b,h,d,s) f16
//   attn out: (b,s,hid) f16
// Two-pass streaming softmax with wave-uniform running max.
// ---------------------------------------------------------------------------
__global__ void k_attn(const _Float16* __restrict__ Qp,
                       const _Float16* __restrict__ Kp,
                       const _Float16* __restrict__ Vt,
                       const int* __restrict__ vlen,
                       _Float16* __restrict__ attn) {
  __shared__ alignas(32) _Float16 lds[16 * 40];   // 16 rows x 32 cols, pad to 40

  const int lane = threadIdx.x & 31;
  const int l16 = lane & 15;
  const int hlf = lane >> 4;
  const int qt = blockIdx.x & (S_ / 16 - 1);
  const int bh = blockIdx.x >> 7;                 // / (S_/16)
  const int h  = bh & (NH_ - 1);
  const int b  = bh >> 3;                         // / NH_
  const int vl = vlen[b];
  const _Float16* q  = Qp + (size_t)bh * S_ * DH_;
  const _Float16* kp = Kp + (size_t)bh * S_ * DH_;
  const _Float16* vt = Vt + (size_t)bh * DH_ * S_;
  const int q0 = qt * 16;

  // Q tile in A-layout, for d-chunks [0,32) and [32,64)
  const _Float16* qrow = q + (size_t)(q0 + l16) * DH_;
  v16h aq0 = cat8(*(const v8h*)(qrow + 16 * hlf),      *(const v8h*)(qrow + 8 + 16 * hlf));
  v16h aq1 = cat8(*(const v8h*)(qrow + 32 + 16 * hlf), *(const v8h*)(qrow + 40 + 16 * hlf));

  float M = -1e30f;
  float l[8];
#pragma unroll
  for (int r = 0; r < 8; ++r) l[r] = 0.0f;

  // ---- pass 1: running max + exp-sum partials ----
#pragma unroll 2
  for (int k0 = 0; k0 < S_; k0 += 16) {
    const _Float16* krow = kp + (size_t)(k0 + l16) * DH_;
    v16h b0 = *(const v16h*)(krow + 16 * hlf);
    v16h b1 = *(const v16h*)(krow + 32 + 16 * hlf);
    v8f s = (v8f){};
    s = wmma_f16(aq0, b0, s);
    s = wmma_f16(aq1, b1, s);
    if (k0 + 16 > vl) {
      const bool om = (k0 + l16) >= vl;
#pragma unroll
      for (int r = 0; r < 8; ++r) s[r] = om ? NEGC : s[r];
    }
    // wave-uniform tile max
    float t = s[0];
#pragma unroll
    for (int r = 1; r < 8; ++r) t = fmaxf(t, s[r]);
    t = fmaxf(t, __shfl_xor(t, 1));
    t = fmaxf(t, __shfl_xor(t, 2));
    t = fmaxf(t, __shfl_xor(t, 4));
    t = fmaxf(t, __shfl_xor(t, 8));
    t = fmaxf(t, __shfl_xor(t, 16));
    if (t > M) {
      const float c = __expf(M - t);
#pragma unroll
      for (int r = 0; r < 8; ++r) l[r] *= c;
      M = t;
    }
#pragma unroll
    for (int r = 0; r < 8; ++r) l[r] += __expf(s[r] - M);
  }

  // finalize row sums (reduce across the 16-lane half; xor 1/2/4/8 stays in half)
  float rinv[8];
#pragma unroll
  for (int r = 0; r < 8; ++r) {
    float t = l[r];
    t += __shfl_xor(t, 1);
    t += __shfl_xor(t, 2);
    t += __shfl_xor(t, 4);
    t += __shfl_xor(t, 8);
    rinv[r] = 1.0f / t;
  }

  // ---- pass 2: P = softmax tile -> LDS (D-layout -> A-layout), then P*V ----
  v8f oa[4];
#pragma unroll
  for (int t = 0; t < 4; ++t) oa[t] = (v8f){};

  for (int k0 = 0; k0 < S_; k0 += 32) {
#pragma unroll
    for (int st = 0; st < 2; ++st) {
      const int ks = k0 + st * 16;
      const _Float16* krow = kp + (size_t)(ks + l16) * DH_;
      v16h b0 = *(const v16h*)(krow + 16 * hlf);
      v16h b1 = *(const v16h*)(krow + 32 + 16 * hlf);
      v8f s = (v8f){};
      s = wmma_f16(aq0, b0, s);
      s = wmma_f16(aq1, b1, s);
      const bool om = (ks + l16) >= vl;
#pragma unroll
      for (int r = 0; r < 8; ++r) {
        const float sv = om ? NEGC : s[r];
        const float p = __expf(sv - M) * rinv[r];
        lds[(r + 8 * hlf) * 40 + st * 16 + l16] = (_Float16)p;
      }
    }
    __syncthreads();   // cross-lane RAW through LDS (single-wave WG: cheap)
    v16h pa = cat8(*(const v8h*)&lds[l16 * 40 + 16 * hlf],
                   *(const v8h*)&lds[l16 * 40 + 8 + 16 * hlf]);
#pragma unroll
    for (int t = 0; t < 4; ++t) {
      // Vt row (= dh) is contiguous in k: perfect B-layout 32B loads
      v16h bv = *(const v16h*)(vt + (size_t)(t * 16 + l16) * S_ + k0 + 16 * hlf);
      oa[t] = wmma_f16(pa, bv, oa[t]);
    }
    __syncthreads();   // WAR before next iteration's LDS stores
  }

  // store O tile: (b, s, h*64 + d) f16
#pragma unroll
  for (int t = 0; t < 4; ++t) {
#pragma unroll
    for (int r = 0; r < 8; ++r) {
      const int srow = q0 + r + 8 * hlf;
      attn[((size_t)(b * S_ + srow)) * HID_ + h * DH_ + t * 16 + l16] = (_Float16)oa[t][r];
    }
  }
}

// ---------------------------------------------------------------------------
extern "C" void kernel_launch(void* const* d_in, const int* in_sizes, int n_in,
                              void* d_out, int out_size, void* d_ws, size_t ws_size,
                              hipStream_t stream) {
  (void)in_sizes; (void)n_in; (void)out_size; (void)ws_size;
  const float* xq = (const float*)d_in[0];
  const float* xk = (const float*)d_in[1];
  const float* xv = (const float*)d_in[2];
  const int*   vl = (const int*)d_in[3];
  const float* Wq = (const float*)d_in[4];
  const float* Wk = (const float*)d_in[5];
  const float* Wv = (const float*)d_in[6];
  const float* Wo = (const float*)d_in[7];
  float* out = (float*)d_out;

  char* w = (char*)d_ws;
  size_t off = 0;
  auto alloc = [&](size_t bytes) -> void* {
    void* p = w + off;
    off = (off + bytes + 255) & ~(size_t)255;
    return p;
  };
  const size_t szX = (size_t)MROWS * HID_ * sizeof(_Float16);   // 8 MB
  const size_t szW = (size_t)HID_ * HID_ * sizeof(_Float16);    // 0.5 MB

  _Float16* xq16 = (_Float16*)alloc(szX);
  _Float16* xk16 = (_Float16*)alloc(szX);
  _Float16* xv16 = (_Float16*)alloc(szX);
  _Float16* wq16 = (_Float16*)alloc(szW);
  _Float16* wk16 = (_Float16*)alloc(szW);
  _Float16* wv16 = (_Float16*)alloc(szW);
  _Float16* wo16 = (_Float16*)alloc(szW);
  _Float16* Qp   = (_Float16*)alloc(szX);
  _Float16* Kp   = (_Float16*)alloc(szX);
  // Aliased scratch (stream-order makes this safe):
  _Float16* Vt     = xk16;   // xk16 dead after K projection
  _Float16* attn16 = xq16;   // xq16 dead after Q projection

  const int nX = MROWS * HID_;
  const int nW = HID_ * HID_;
  k_cvt<<<(nX + 255) / 256, 256, 0, stream>>>(xq, xq16, nX);
  k_cvt<<<(nX + 255) / 256, 256, 0, stream>>>(xk, xk16, nX);
  k_cvt<<<(nX + 255) / 256, 256, 0, stream>>>(xv, xv16, nX);
  k_cvt<<<(nW + 255) / 256, 256, 0, stream>>>(Wq, wq16, nW);
  k_cvt<<<(nW + 255) / 256, 256, 0, stream>>>(Wk, wk16, nW);
  k_cvt<<<(nW + 255) / 256, 256, 0, stream>>>(Wv, wv16, nW);
  k_cvt<<<(nW + 255) / 256, 256, 0, stream>>>(Wo, wo16, nW);

  dim3 gg(MROWS / 64, HID_ / 64);   // (128, 8)
  k_gemm_xwt<<<gg, 128, 0, stream>>>(xq16, wq16, 0, 0.125f, Qp, nullptr);   // 1/sqrt(64)
  k_gemm_xwt<<<gg, 128, 0, stream>>>(xk16, wk16, 1, 1.0f,   Kp, nullptr);
  k_gemm_xwt<<<gg, 128, 0, stream>>>(xv16, wv16, 2, 1.0f,   Vt, nullptr);

  k_attn<<<B_ * NH_ * (S_ / 16), 32, 0, stream>>>(Qp, Kp, Vt, vl, attn16);

  k_gemm_xwt<<<gg, 128, 0, stream>>>(attn16, wo16, 3, 1.0f, nullptr, out);
}
